// noisy_top_k_gating_84688165142881
// MI455X (gfx1250) — matrix operand order repaired and checked by reading, not compiled, not run on hardware
//
#include <hip/hip_runtime.h>
#include <hip/hip_bf16.h>
#include <math.h>

// Problem constants (match reference)
#define B_ROWS   262144
#define DIM      256
#define NEXP     64
#define WAVES    8
#define ROWS_PER_WG (16 * WAVES)          // 128 rows per workgroup
#define NBLK     (B_ROWS / ROWS_PER_WG)   // 2048 row blocks
#define GRID     1024                     // each WG handles exactly 2 blocks
#define WSTRIDE  264                      // padded bf16 row stride (16B aligned, bank-spread)
#define SSTRIDE  65                       // padded fp32 scratch stride (conflict-free)
#define NOISE_EPS 0.01f

typedef __attribute__((ext_vector_type(16))) __bf16          v16bf;
typedef __attribute__((ext_vector_type(2)))  __bf16          v2bf;
typedef __attribute__((ext_vector_type(8)))  unsigned short  v8u;
typedef __attribute__((ext_vector_type(8)))  unsigned int    v8ui;
typedef __attribute__((ext_vector_type(16))) unsigned short  v16u;
typedef __attribute__((ext_vector_type(8)))  float           v8f;
typedef __attribute__((ext_vector_type(4)))  float           v4f;

// Packed f32x2 -> bf16x2 (single v_cvt_pk_bf16_f32 when available)
#if __has_builtin(__builtin_amdgcn_cvt_pk_bf16_f32)
__device__ __forceinline__ unsigned int pk_bf16(float a, float b) {
    v2bf r = __builtin_amdgcn_cvt_pk_bf16_f32(a, b);
    return __builtin_bit_cast(unsigned int, r);
}
#else
__device__ __forceinline__ unsigned int pk_bf16(float a, float b) {
    v2bf r;
    r[0] = (__bf16)a;
    r[1] = (__bf16)b;
    return __builtin_bit_cast(unsigned int, r);
}
#endif

__device__ __forceinline__ float softplus_f(float v) {
    return fmaxf(v, 0.0f) + log1pf(__expf(-fabsf(v)));
}

// Load one 32x16 bf16 B-fragment: lane n holds expert col (n%16), K = 16*(n/16)+0..15,
// which is 16 contiguous bf16 of one weight row in LDS -> two b128 DS loads.
__device__ __forceinline__ v16bf ld_bfrag(const unsigned short* p) {
    v8u lo = *(const v8u*)p;
    v8u hi = *(const v8u*)(p + 8);
    v16u u;
#pragma unroll
    for (int i = 0; i < 8; i++) { u[i] = lo[i]; u[i + 8] = hi[i]; }
    return __builtin_bit_cast(v16bf, u);
}

__global__ void zero_load_kernel(float* __restrict__ load) {
    load[threadIdx.x] = 0.0f;
}

__global__ __launch_bounds__(256) void noisy_topk_gating_kernel(
    const float* __restrict__ x,
    const float* __restrict__ w_gate,
    const float* __restrict__ w_noise,
    float* __restrict__ load,      // [64]
    float* __restrict__ gates)     // [B, 64]
{
    __shared__ __align__(16) unsigned short s_w[2 * NEXP * WSTRIDE]; // bf16 w_gate | w_noise
    __shared__ float s_scr[WAVES][2][16][SSTRIDE];                   // per-wave clean / stddev
    __shared__ float s_part[WAVES][16][2][4];                        // per-half top3 + partial sum
    __shared__ float s_row[WAVES][16][4];                            // t0,t1,t2,sumexp per row
    __shared__ float s_load[NEXP];

    const int tid  = threadIdx.x;
    const int wave = tid >> 5;
    const int lane = tid & 31;
    const int half = lane >> 4;
    const int l16  = lane & 15;

    if (tid < NEXP) s_load[tid] = 0.0f;

    // ---- Stage both weight matrices into LDS as bf16 (coalesced float4 reads) ----
    for (int i = tid; i < 2 * NEXP * DIM / 4; i += 256) {
        int mat = i / (NEXP * DIM / 4);
        int rem = i % (NEXP * DIM / 4);
        int row = rem / (DIM / 4);
        int c4  = rem % (DIM / 4);
        const float* src = mat ? w_noise : w_gate;
        v4f v = *(const v4f*)(src + row * DIM + c4 * 4);
        unsigned int* d = (unsigned int*)&s_w[(mat * NEXP + row) * WSTRIDE + c4 * 4];
        d[0] = pk_bf16(v.x, v.y);
        d[1] = pk_bf16(v.z, v.w);
    }
    __syncthreads();

    float lacc0 = 0.0f, lacc1 = 0.0f;   // per-lane load partials: expert lane, lane+32

    for (int blk = blockIdx.x; blk < NBLK; blk += GRID) {
        const int rowBase = blk * ROWS_PER_WG + wave * 16;
        const int myRow   = rowBase + l16;

        v8f accC[4], accN[4];
#pragma unroll
        for (int t = 0; t < 4; t++) {
            accC[t] = (v8f){0.f,0.f,0.f,0.f,0.f,0.f,0.f,0.f};
            accN[t] = (v8f){0.f,0.f,0.f,0.f,0.f,0.f,0.f,0.f};
        }

        // ---- Dual GEMM: clean = x*w_gate^T, noiseRaw = x*w_noise^T via WMMA bf16 ----
        const float* xrow = x + (size_t)myRow * DIM;
#pragma unroll
        for (int ks = 0; ks < DIM / 32; ks++) {
            const int k0 = ks * 32;
            // A fragment (16x32 bf16): lane holds row l16, K = 16*half+0..15
            v4f a0 = *(const v4f*)(xrow + k0 + 16 * half);
            v4f a1 = *(const v4f*)(xrow + k0 + 16 * half + 4);
            v4f a2 = *(const v4f*)(xrow + k0 + 16 * half + 8);
            v4f a3 = *(const v4f*)(xrow + k0 + 16 * half + 12);
            v8ui u;
            u[0] = pk_bf16(a0.x, a0.y);  u[1] = pk_bf16(a0.z, a0.w);
            u[2] = pk_bf16(a1.x, a1.y);  u[3] = pk_bf16(a1.z, a1.w);
            u[4] = pk_bf16(a2.x, a2.y);  u[5] = pk_bf16(a2.z, a2.w);
            u[6] = pk_bf16(a3.x, a3.y);  u[7] = pk_bf16(a3.z, a3.w);
            v16bf av = __builtin_bit_cast(v16bf, u);

#pragma unroll
            for (int t = 0; t < 4; t++) {
                const int wrow = t * 16 + l16;
                v16bf bg = ld_bfrag(&s_w[(0 * NEXP + wrow) * WSTRIDE + k0 + 16 * half]);
                v16bf bn = ld_bfrag(&s_w[(1 * NEXP + wrow) * WSTRIDE + k0 + 16 * half]);
                accC[t] = __builtin_amdgcn_wmma_f32_16x16x32_bf16(
                    false, av, false, bg, (short)0, accC[t], false, false);
                accN[t] = __builtin_amdgcn_wmma_f32_16x16x32_bf16(
                    false, av, false, bn, (short)0, accN[t], false, false);
            }
        }

        // ---- Prefetch next block's x tile while we run the epilogue ----
        if (blk + GRID < NBLK) {
            const float* xn = x + ((size_t)(blk + GRID) * ROWS_PER_WG + wave * 16) * DIM;
#pragma unroll
            for (int r = 0; r < 16; r++)
                __builtin_prefetch(xn + r * DIM + lane * 8, 0, 0);
        }

        // ---- Spill: clean and stddev = softplus(noiseRaw)+eps  (D layout: row=r+8*half, col=l16) ----
#pragma unroll
        for (int t = 0; t < 4; t++) {
#pragma unroll
            for (int r = 0; r < 8; r++) {
                const int rr = r + 8 * half;
                const int e  = t * 16 + l16;
                s_scr[wave][0][rr][e] = accC[t][r];
                s_scr[wave][1][rr][e] = softplus_f(accN[t][r]) + NOISE_EPS;
            }
        }
        __syncthreads();

        // ---- Top-3 scan, full wave: lane handles row l16, expert half `half` (32 experts) ----
        {
            const float* pc = &s_scr[wave][0][l16][32 * half];
            const float* ps = &s_scr[wave][1][l16][32 * half];
            float a0 = -INFINITY, a1 = -INFINITY, a2 = -INFINITY;
            for (int i = 0; i < 32; i++) {
                float h = pc[i] + ps[i];
                if (h > a0)      { a2 = a1; a1 = a0; a0 = h; }
                else if (h > a1) { a2 = a1; a1 = h; }
                else if (h > a2) { a2 = h; }
            }
            s_part[wave][l16][half][0] = a0;
            s_part[wave][l16][half][1] = a1;
            s_part[wave][l16][half][2] = a2;
        }
        __syncthreads();

        // ---- Merge the two sorted triples per row (lanes 0..15) ----
        if (lane < 16) {
            float x0 = s_part[wave][lane][0][0], x1 = s_part[wave][lane][0][1], x2 = s_part[wave][lane][0][2];
            float y0 = s_part[wave][lane][1][0], y1 = s_part[wave][lane][1][1], y2 = s_part[wave][lane][1][2];
            float t0, t1, t2;
            if (x0 >= y0) {
                t0 = x0;
                if (x1 >= y0) { t1 = x1; t2 = fmaxf(x2, y0); }
                else          { t1 = y0; t2 = fmaxf(x1, y1); }
            } else {
                t0 = y0;
                if (y1 >= x0) { t1 = y1; t2 = fmaxf(y2, x0); }
                else          { t1 = x0; t2 = fmaxf(x1, y1); }
            }
            s_row[wave][lane][0] = t0; s_row[wave][lane][1] = t1; s_row[wave][lane][2] = t2;
        }
        __syncthreads();

        // ---- Softmax denominator, full wave (tie-exact: sum over h >= thr_k) ----
        {
            const float t0 = s_row[wave][l16][0];
            const float t1 = s_row[wave][l16][1];
            const float* pc = &s_scr[wave][0][l16][32 * half];
            const float* ps = &s_scr[wave][1][l16][32 * half];
            float sum = 0.0f;
            for (int i = 0; i < 32; i++) {
                float h = pc[i] + ps[i];
                if (h >= t1) sum += __expf(h - t0);
            }
            s_part[wave][l16][half][3] = sum;
        }
        __syncthreads();
        if (lane < 16)
            s_row[wave][lane][3] = s_part[wave][lane][0][3] + s_part[wave][lane][1][3];
        __syncthreads();

        // ---- Gates (coalesced, lane = expert) + load partials (normal_cdf) ----
        for (int r = 0; r < 16; r++) {
            const float t0 = s_row[wave][r][0];
            const float t1 = s_row[wave][r][1];   // thr_k (top-2 threshold)
            const float t2 = s_row[wave][r][2];   // thr_in (top-3 value)
            const float invs = __frcp_rn(s_row[wave][r][3]);
            const size_t gbase = (size_t)(rowBase + r) * NEXP;
#pragma unroll
            for (int hh = 0; hh < 2; hh++) {
                const int e = lane + hh * 32;
                float c = s_scr[wave][0][r][e];
                float s = s_scr[wave][1][r][e];
                float h = c + s;
                gates[gbase + e] = (h >= t1) ? __expf(h - t0) * invs : 0.0f;
                float thr = (h > t2) ? t2 : t1;
                float p = 0.5f * (1.0f + erff(__fdividef(c - thr, 1.41421356237f * s + 1e-20f)));
                if (hh == 0) lacc0 += p; else lacc1 += p;
            }
        }
        __syncthreads();   // protect s_scr/s_row/s_part before next block iteration
    }

    // ---- Reduce load: LDS atomics, then 64 global atomics per workgroup ----
    atomicAdd(&s_load[lane],      lacc0);
    atomicAdd(&s_load[lane + 32], lacc1);
    __syncthreads();
    if (tid < NEXP) atomicAdd(&load[tid], s_load[tid]);
}

extern "C" void kernel_launch(void* const* d_in, const int* in_sizes, int n_in,
                              void* d_out, int out_size, void* d_ws, size_t ws_size,
                              hipStream_t stream) {
    (void)in_sizes; (void)n_in; (void)out_size; (void)d_ws; (void)ws_size;
    const float* x       = (const float*)d_in[0];
    const float* w_gate  = (const float*)d_in[1];
    const float* w_noise = (const float*)d_in[2];
    float* load  = (float*)d_out;          // first 64 floats
    float* gates = (float*)d_out + NEXP;   // then B*E gates

    zero_load_kernel<<<1, NEXP, 0, stream>>>(load);
    noisy_topk_gating_kernel<<<GRID, 256, 0, stream>>>(x, w_gate, w_noise, load, gates);
}
